// ElmanSilu_12137577578505
// MI455X (gfx1250) — matrix operand order repaired and checked by reading, not compiled, not run on hardware
//
#include <hip/hip_runtime.h>
#include <math.h>

// ---------------- problem constants ----------------
#define T_STEPS   1024
#define BATCH     32
#define D_IN      1024
#define D_H       1024
#define GATE      2048              // 2*D_H
#define M_TOTAL   (T_STEPS*BATCH)   // 32768
#define NWG_SCAN  16

// ---------------- workspace layout (bytes) ----------------
#define OFF_XW   0ull               // xW fp32: 268435456
#define OFF_XB   268435456ull       // x  bf16:  67108864
#define OFF_WXB  335544320ull       // Wx bf16:   4194304
#define OFF_RB   339738624ull       // R  bf16:   4194304
#define OFF_HB   343932928ull       // h  bf16 x2:  131072
#define OFF_CTR  344064000ull       // ctr int:      4096

typedef __attribute__((ext_vector_type(16))) __bf16         v16bf;
typedef __attribute__((ext_vector_type(8)))  float          v8f;
typedef __attribute__((ext_vector_type(4)))  float          v4f;
typedef __attribute__((ext_vector_type(4)))  unsigned short v4us;
typedef __attribute__((ext_vector_type(4)))  unsigned       v4u;
typedef __attribute__((ext_vector_type(8)))  int            v8i;
typedef __attribute__((ext_vector_type(4)))  int            v4i;

// LDS tile geometry: 32 bf16 per row + 8 bf16 pad (TDM pad: 4 DWORDs per 16 DWORDs)
#define LDS_ROW   40                // elements per padded row (80 bytes)

__device__ __forceinline__ v8f wmma_bf16(v16bf a, v16bf b, v8f c) {
    return __builtin_amdgcn_wmma_f32_16x16x32_bf16(false, a, false, b,
                                                   (short)0, c, false, false);
}

__device__ __forceinline__ unsigned lds_off(const void* p) {
    // generic pointer to __shared__: low 32 bits are the LDS byte offset
    return (unsigned)(uintptr_t)p;
}

// ---------------- TDM: load a 128-row x 32-col bf16 tile into LDS -------------
// D# per ISA 8.3-8.5: count=1, type=2("image"), data_size=2B, pad_enable,
// pad_interval=16 DWORDs (code 3), pad_amount=4 DWORDs (code 3),
// tile_dim0=32, tile_dim1=128, tensor_dim0=1024, tensor_dim0_stride=1024.
__device__ __forceinline__ void tdm_load_tile(const __bf16* gbase,
                                              unsigned rows_total,
                                              unsigned row0, unsigned k0,
                                              unsigned ldsbyte) {
    unsigned long long ga = (unsigned long long)(uintptr_t)gbase +
                            ((unsigned long long)row0 * 1024ull + k0) * 2ull;
    v4u g0 = { 1u,                                    // count=1, user mode
               ldsbyte,                               // lds_addr
               (unsigned)ga,                          // global_addr lo
               (unsigned)((ga >> 32) & 0x01FFFFFFu) | 0x80000000u }; // hi | type=2
    v8i g1 = { (int)((1u << 16) | (1u << 20) | (3u << 22) | (3u << 25)),
               (int)(1024u << 16),                      // tensor_dim0 = 1024
               (int)((rows_total & 0xFFFFu) << 16),     // tensor_dim1 lo16
               (int)((rows_total >> 16) | (32u << 16)), // tensor_dim1 hi | tile_dim0=32
               (int)128,                                // tile_dim1=128
               (int)1024,                               // tensor_dim0_stride = 1024
               0, 0 };
    v4i gz = { 0, 0, 0, 0 };
#if defined(__clang_major__) && (__clang_major__ >= 23)
    v8i gz8 = { 0, 0, 0, 0, 0, 0, 0, 0 };
    __builtin_amdgcn_tensor_load_to_lds(g0, g1, gz, gz, gz8, 0);
#else
    __builtin_amdgcn_tensor_load_to_lds(g0, g1, gz, gz, 0);
#endif
}

// ---------------- LDS fragment loaders (padded rows: bank-conflict free) ------
__device__ __forceinline__ v16bf lds_frag_a(const __bf16* buf, int mm, int lane) {
    int r  = mm + (lane & 15);
    int kb = (lane >> 4) << 3;                 // +8 for upper half-wave
    const __bf16* p = buf + r * LDS_ROW + kb;
    union { v16bf v; uint4 q[2]; } f;
    f.q[0] = *(const uint4*)(p);
    f.q[1] = *(const uint4*)(p + 16);
    return f.v;
}
__device__ __forceinline__ v16bf lds_frag_b(const __bf16* buf, int nn, int lane) {
    int r  = nn + (lane & 15);
    int kb = (lane >> 4) << 3;                 // +8 for upper half-wave
    const __bf16* p = buf + r * LDS_ROW + kb;
    union { v16bf v; uint4 q[2]; } f;
    f.q[0] = *(const uint4*)(p);
    f.q[1] = *(const uint4*)(p + 16);
    return f.v;
}

// ---------------- global fragment loaders (used by scan kernel) ---------------
__device__ __forceinline__ v16bf load_frag_a(const __bf16* __restrict__ base,
                                             int ld, int m0, int k0, int lane) {
    int r  = m0 + (lane & 15);
    int kb = k0 + ((lane >> 4) << 3);
    const __bf16* p = base + (size_t)r * ld + kb;
    union { v16bf v; uint4 q[2]; } f;
    f.q[0] = *(const uint4*)(p);
    f.q[1] = *(const uint4*)(p + 16);
    return f.v;
}
__device__ __forceinline__ v16bf load_frag_b(const __bf16* __restrict__ base,
                                             int ld, int n0, int k0, int lane) {
    int r  = n0 + (lane & 15);
    int kb = k0 + ((lane >> 4) << 3);
    const __bf16* p = base + (size_t)r * ld + kb;
    union { v16bf v; uint4 q[2]; } f;
    f.q[0] = *(const uint4*)(p);
    f.q[1] = *(const uint4*)(p + 16);
    return f.v;
}

// ---------------- fp32 -> bf16 convert (vectorized, NT source reads) ----------
__global__ void cvt_f32_bf16_kernel(const v4f* __restrict__ in,
                                    v4us* __restrict__ out, size_t n4) {
    size_t i = (size_t)blockIdx.x * blockDim.x + threadIdx.x;
    size_t stride = (size_t)gridDim.x * blockDim.x;
    for (; i < n4; i += stride) {
        v4f f = __builtin_nontemporal_load(in + i);
        v4us o;
        o.x = __builtin_bit_cast(unsigned short, (__bf16)f.x);
        o.y = __builtin_bit_cast(unsigned short, (__bf16)f.y);
        o.z = __builtin_bit_cast(unsigned short, (__bf16)f.z);
        o.w = __builtin_bit_cast(unsigned short, (__bf16)f.w);
        out[i] = o;
    }
}

__global__ void init_scan_kernel(const float* __restrict__ h0,
                                 __bf16* __restrict__ hbuf,
                                 int* __restrict__ ctr) {
    int i = blockIdx.x * blockDim.x + threadIdx.x;
    if (i < T_STEPS) ctr[i] = 0;
    for (int idx = i; idx < BATCH * D_H; idx += gridDim.x * blockDim.x)
        hbuf[idx] = (__bf16)h0[idx];
}

// ---------------- phase 1: xW = x @ Wx^T + bias --------------------------------
// WG tile 128x128, K-step 32, TDM double-buffered LDS staging.
// 8 waves; wave w computes rows [mw, mw+32) x cols [nw, nw+64): 2x4 WMMA tiles.
__global__ __launch_bounds__(256, 1)
void gemm_xw_kernel(const __bf16* __restrict__ xb,
                    const __bf16* __restrict__ wxb,
                    const float*  __restrict__ bias,
                    float* __restrict__ xW) {
    __shared__ __align__(16) __bf16 sA[2][128 * LDS_ROW];
    __shared__ __align__(16) __bf16 sB[2][128 * LDS_ROW];

    int wave = threadIdx.x >> 5, lane = threadIdx.x & 31;
    int m0 = blockIdx.x * 128;
    int n0 = blockIdx.y * 128;
    int mw = (wave >> 1) * 32;      // 0,32,64,96
    int nw = (wave & 1) * 64;       // 0,64
    bool issuer = (wave == 0);

    v8f zero = {0.f,0.f,0.f,0.f,0.f,0.f,0.f,0.f};
    v8f acc[2][4];
#pragma unroll
    for (int mi = 0; mi < 2; ++mi)
#pragma unroll
        for (int ni = 0; ni < 4; ++ni) acc[mi][ni] = zero;

    if (issuer) {
        tdm_load_tile(xb,  M_TOTAL, (unsigned)m0, 0, lds_off(&sA[0][0]));
        tdm_load_tile(wxb, GATE,    (unsigned)n0, 0, lds_off(&sB[0][0]));
    }

    for (int j = 0; j < 32; ++j) {          // K = 32 steps of 32
        if (issuer) __builtin_amdgcn_s_wait_tensorcnt(0);
        __syncthreads();                    // front buffer ready; back buffer free
        if (issuer && (j + 1 < 32)) {
            int nb = (j + 1) & 1;
            tdm_load_tile(xb,  M_TOTAL, (unsigned)m0, (unsigned)((j + 1) * 32),
                          lds_off(&sA[nb][0]));
            tdm_load_tile(wxb, GATE,    (unsigned)n0, (unsigned)((j + 1) * 32),
                          lds_off(&sB[nb][0]));
        }
        int cb = j & 1;
        // hoist ALL fragment loads so the scheduler overlaps them with WMMA
        v16bf afrag[2], bfrag[4];
#pragma unroll
        for (int mi = 0; mi < 2; ++mi)
            afrag[mi] = lds_frag_a(&sA[cb][0], mw + mi * 16, lane);
#pragma unroll
        for (int ni = 0; ni < 4; ++ni)
            bfrag[ni] = lds_frag_b(&sB[cb][0], nw + ni * 16, lane);
#pragma unroll
        for (int mi = 0; mi < 2; ++mi)
#pragma unroll
            for (int ni = 0; ni < 4; ++ni)
                acc[mi][ni] = wmma_bf16(afrag[mi], bfrag[ni], acc[mi][ni]);
    }

    // epilogue: + bias, store fp32 xW
    int mbase0 = m0 + mw + ((lane >> 4) << 3);
#pragma unroll
    for (int mi = 0; mi < 2; ++mi) {
        int mb = mbase0 + mi * 16;
#pragma unroll
        for (int ni = 0; ni < 4; ++ni) {
            int col = n0 + nw + ni * 16 + (lane & 15);
            float bb = bias[col];
#pragma unroll
            for (int r = 0; r < 8; ++r)
                xW[(size_t)(mb + r) * GATE + col] = acc[mi][ni][r] + bb;
        }
    }
}

// ---------------- phase 2: persistent sequential scan --------------------------
__global__ __launch_bounds__(256, 1)
void scan_kernel(const float*  __restrict__ xW,
                 const __bf16* __restrict__ Rb,
                 __bf16*       __restrict__ hbuf,   // 2 x [BATCH*D_H]
                 float*        __restrict__ hs,     // output (T,B,D_H)
                 int*          __restrict__ ctr) {
    int wave = threadIdx.x >> 5, lane = threadIdx.x & 31;
    int task = blockIdx.x * 8 + wave;          // 0..127
    int m0   = (task & 1) * 16;
    int jn   = task >> 1;                      // 0..63
    int ccol = jn * 16;
    int gcol = D_H + jn * 16;
    int col  = lane & 15;
    int mbase = m0 + ((lane >> 4) << 3);

    // software-pipelined xW seed loads: pre-load step 0, then always fetch t+1
    // during step t so the HBM latency hides under the K-loop + barrier.
    v8f pre_c, pre_g;
#pragma unroll
    for (int r = 0; r < 8; ++r) {
        pre_c[r] = xW[(size_t)(mbase + r) * GATE + (ccol + col)];
        pre_g[r] = xW[(size_t)(mbase + r) * GATE + (gcol + col)];
    }

    for (int t = 0; t < T_STEPS; ++t) {
        const __bf16* hin  = hbuf + (size_t)(t & 1) * (BATCH * D_H);
        __bf16*       hout = hbuf + (size_t)((t + 1) & 1) * (BATCH * D_H);

        v8f acc_c = pre_c, acc_g = pre_g;
        if (t + 1 < T_STEPS) {
            const float* xwn = xW + (size_t)(t + 1) * BATCH * GATE;
#pragma unroll
            for (int r = 0; r < 8; ++r) {
                pre_c[r] = xwn[(size_t)(mbase + r) * GATE + (ccol + col)];
                pre_g[r] = xwn[(size_t)(mbase + r) * GATE + (gcol + col)];
            }
        }

#pragma unroll 4
        for (int k0 = 0; k0 < D_H; k0 += 32) {
            v16bf a  = load_frag_a(hin, D_H, m0, k0, lane);
            v16bf bc = load_frag_b(Rb, D_H, ccol, k0, lane);
            v16bf bg = load_frag_b(Rb, D_H, gcol, k0, lane);
            acc_c = wmma_bf16(a, bc, acc_c);
            acc_g = wmma_bf16(a, bg, acc_g);
        }

        float* hst = hs + (size_t)t * BATCH * D_H;
#pragma unroll
        for (int r = 0; r < 8; ++r) {
            float c = acc_c[r], g = acc_g[r];
            float hn = tanhf(c) * (g / (1.0f + __expf(-g)));
            int row = mbase + r;
            __builtin_nontemporal_store(hn, &hst[(size_t)row * D_H + (ccol + col)]);
            hout[(size_t)row * D_H + (ccol + col)] = (__bf16)hn;
        }

        __builtin_amdgcn_fence(__ATOMIC_RELEASE, "agent");
        __syncthreads();
        if (threadIdx.x == 0) {
            __hip_atomic_fetch_add(&ctr[t], 1, __ATOMIC_RELEASE,
                                   __HIP_MEMORY_SCOPE_AGENT);
            while (__hip_atomic_load(&ctr[t], __ATOMIC_ACQUIRE,
                                     __HIP_MEMORY_SCOPE_AGENT) < NWG_SCAN)
                __builtin_amdgcn_s_sleep(2);
        }
        __syncthreads();
        __builtin_amdgcn_fence(__ATOMIC_ACQUIRE, "agent");
    }
}

// ---------------- host launcher ----------------
extern "C" void kernel_launch(void* const* d_in, const int* in_sizes, int n_in,
                              void* d_out, int out_size, void* d_ws, size_t ws_size,
                              hipStream_t stream) {
    const float* x    = (const float*)d_in[0];
    const float* h0   = (const float*)d_in[1];
    const float* Wx   = (const float*)d_in[2];
    const float* R    = (const float*)d_in[3];
    const float* bias = (const float*)d_in[4];
    float* hs = (float*)d_out;

    char*   ws   = (char*)d_ws;
    float*  xW   = (float*)(ws + OFF_XW);
    __bf16* xb   = (__bf16*)(ws + OFF_XB);
    __bf16* wxb  = (__bf16*)(ws + OFF_WXB);
    __bf16* rb   = (__bf16*)(ws + OFF_RB);
    __bf16* hbuf = (__bf16*)(ws + OFF_HB);
    int*    ctr  = (int*)(ws + OFF_CTR);

    cvt_f32_bf16_kernel<<<2048, 256, 0, stream>>>((const v4f*)x,  (v4us*)xb,
                                                  (size_t)M_TOTAL * D_IN / 4);
    cvt_f32_bf16_kernel<<<512, 256, 0, stream>>>((const v4f*)Wx, (v4us*)wxb,
                                                 (size_t)GATE * D_IN / 4);
    cvt_f32_bf16_kernel<<<512, 256, 0, stream>>>((const v4f*)R,  (v4us*)rb,
                                                 (size_t)GATE * D_H / 4);
    init_scan_kernel<<<256, 256, 0, stream>>>(h0, hbuf, ctr);

    gemm_xw_kernel<<<dim3(M_TOTAL / 128, GATE / 128), 256, 0, stream>>>(xb, wxb, bias, xW);
    scan_kernel<<<NWG_SCAN, 256, 0, stream>>>(xW, rb, hbuf, hs, ctr);
}
